// Attention_16793322128104
// MI455X (gfx1250) — matrix operand order repaired and checked by reading, not compiled, not run on hardware
//
#include <hip/hip_runtime.h>

// ---------------------------------------------------------------------------
// MI455X (gfx1250) fused attention block, bf16 WMMA everywhere.
//   dim = 1024, HEADS = 16, d_head = 64, B = 2, N = 2048
// ---------------------------------------------------------------------------

typedef __bf16 bf16_t;
typedef bf16_t v16bf __attribute__((ext_vector_type(16)));
typedef bf16_t v8bf  __attribute__((ext_vector_type(8)));
typedef float  v8f   __attribute__((ext_vector_type(8)));

union AFrag { v16bf v; v8bf h[2]; };

// A tile 16(M) x 32(K), row-major with leading dim `ld` (bf16 elements).
// ISA layout: lane -> m = lane&15, half = lane>>4;
// element e(0..7) -> K = 8*half + e ; e(8..15) -> K = 16 + 8*half + (e-8)
__device__ __forceinline__ AFrag load_a16x32(const bf16_t* base, int ld, int lane) {
  const int m = lane & 15, half = (lane >> 4) & 1;
  AFrag f;
  const bf16_t* p = base + m * ld + 8 * half;
  f.h[0] = *reinterpret_cast<const v8bf*>(p);
  f.h[1] = *reinterpret_cast<const v8bf*>(p + 16);
  return f;
}

// B tile 32(K) x 16(N), stored N-major (i.e. element (k,n) at base + n*ld + k).
// ISA layout: lane -> n = lane&15, half = lane>>4;
// element e(0..7) -> K = 8*half + e ; e(8..15) -> K = 16 + 8*half + (e-8)
__device__ __forceinline__ AFrag load_b32x16(const bf16_t* base, int ld, int lane) {
  const int n = lane & 15, half = (lane >> 4) & 1;
  AFrag f;
  const bf16_t* p = base + n * ld + 8 * half;
  f.h[0] = *reinterpret_cast<const v8bf*>(p);
  f.h[1] = *reinterpret_cast<const v8bf*>(p + 16);
  return f;
}

__device__ __forceinline__ v8f wmma_bf16(const AFrag& a, const AFrag& b, v8f c) {
  return __builtin_amdgcn_wmma_f32_16x16x32_bf16(
      /*neg_a=*/false, a.v, /*neg_b=*/false, b.v,
      /*c_mod=*/(short)0, c, /*reuse_a=*/false, /*reuse_b=*/false);
}

__device__ __forceinline__ float hmax16(float v) {
  v = fmaxf(v, __shfl_xor(v, 1, 32));
  v = fmaxf(v, __shfl_xor(v, 2, 32));
  v = fmaxf(v, __shfl_xor(v, 4, 32));
  v = fmaxf(v, __shfl_xor(v, 8, 32));
  return v;
}
__device__ __forceinline__ float hsum16(float v) {
  v += __shfl_xor(v, 1, 32);
  v += __shfl_xor(v, 2, 32);
  v += __shfl_xor(v, 4, 32);
  v += __shfl_xor(v, 8, 32);
  return v;
}

// ---------------------------------------------------------------------------
// f32 -> bf16 convert / transpose (one-shot data prep)
// ---------------------------------------------------------------------------
__global__ void __launch_bounds__(256) cvt_bf16(const float* __restrict__ src,
                                                bf16_t* __restrict__ dst, int n) {
  int i = blockIdx.x * 256 + threadIdx.x;
  if (i < n) dst[i] = (bf16_t)src[i];
}

// src [K x N] row-major -> dst [N x K] (N-major weights for contiguous B frags)
__global__ void __launch_bounds__(256) cvt_bf16_T(const float* __restrict__ src,
                                                  bf16_t* __restrict__ dst,
                                                  int K, int N) {
  int i = blockIdx.x * 256 + threadIdx.x;
  if (i < K * N) {
    int n = i / K, k = i - n * K;
    dst[i] = (bf16_t)src[k * N + n];
  }
}

// ---------------------------------------------------------------------------
// GEMM 1: qkv = x @ Wqkv + bqkv  (M=4096, K=1024, Ntot=3072)
// wave tile 32x64; epilogue scatters into per-head q / k / v-transposed bufs.
// grid (128, 12), block 128 (4 waves)
// ---------------------------------------------------------------------------
__global__ void __launch_bounds__(128) gemm_qkv(const bf16_t* __restrict__ xb,
                                                const bf16_t* __restrict__ wT,
                                                const float* __restrict__ bias,
                                                bf16_t* __restrict__ qb,
                                                bf16_t* __restrict__ kb,
                                                bf16_t* __restrict__ vtb) {
  const int lane = threadIdx.x & 31;
  const int w    = threadIdx.x >> 5;
  const int r0   = blockIdx.x * 32;
  const int c0   = blockIdx.y * 256 + w * 64;

  v8f acc[2][4];
#pragma unroll
  for (int mt = 0; mt < 2; ++mt)
#pragma unroll
    for (int nt = 0; nt < 4; ++nt)
#pragma unroll
      for (int i = 0; i < 8; ++i) acc[mt][nt][i] = 0.f;

  for (int k0 = 0; k0 < 1024; k0 += 32) {
    AFrag a0 = load_a16x32(xb + r0 * 1024 + k0, 1024, lane);
    AFrag a1 = load_a16x32(xb + (r0 + 16) * 1024 + k0, 1024, lane);
#pragma unroll
    for (int nt = 0; nt < 4; ++nt) {
      AFrag b = load_b32x16(wT + (c0 + 16 * nt) * 1024 + k0, 1024, lane);
      acc[0][nt] = wmma_bf16(a0, b, acc[0][nt]);
      acc[1][nt] = wmma_bf16(a1, b, acc[1][nt]);
    }
  }

  const int half = (lane >> 4) & 1, n = lane & 15;
#pragma unroll
  for (int mt = 0; mt < 2; ++mt)
#pragma unroll
    for (int nt = 0; nt < 4; ++nt) {
      const int col = c0 + 16 * nt + n;
      const float bv = bias[col];
      const int which = col >> 10;        // 0=q 1=k 2=v
      const int cc = col & 1023;
      const int h = cc >> 6, d = cc & 63;
#pragma unroll
      for (int i = 0; i < 8; ++i) {
        const int row = r0 + 16 * mt + i + 8 * half;  // global token row
        const int b   = row >> 11, ns = row & 2047;
        const int bh  = b * 16 + h;
        const bf16_t val = (bf16_t)(acc[mt][nt][i] + bv);
        if (which == 0)      qb [(bh * 2048 + ns) * 64 + d] = val;
        else if (which == 1) kb [(bh * 2048 + ns) * 64 + d] = val;
        else                 vtb[(bh * 64 + d) * 2048 + ns] = val;  // transposed
      }
    }
}

// ---------------------------------------------------------------------------
// Flash attention: per (bh, 64-row block). block = 4 waves, wave = 16 rows.
// Online softmax, 32-key blocks; S via 4 WMMAs, P·Vt via 4 WMMAs per block.
// grid 32*32 = 1024, block 128
// ---------------------------------------------------------------------------
__global__ void __launch_bounds__(128) attn_fa(const bf16_t* __restrict__ qb,
                                               const bf16_t* __restrict__ kb,
                                               const bf16_t* __restrict__ vtb,
                                               bf16_t* __restrict__ ab,
                                               float scale) {
  __shared__ bf16_t psh[4][16][40];   // per-wave 16x32 P tile, padded rows (80B, 16B-aligned)

  const int lane = threadIdx.x & 31;
  const int w    = threadIdx.x >> 5;
  const int half = (lane >> 4) & 1, n = lane & 15;
  const int bh   = blockIdx.x >> 5;
  const int rb   = blockIdx.x & 31;
  const int r0   = rb * 64 + w * 16;          // query rows r0..r0+15

  const bf16_t* qrow = qb + (bh * 2048 + r0) * 64;
  const AFrag qa0 = load_a16x32(qrow, 64, lane);       // d 0..31
  const AFrag qa1 = load_a16x32(qrow + 32, 64, lane);  // d 32..63

  float mrow[8], lrow[8];
  v8f o[4];
#pragma unroll
  for (int i = 0; i < 8; ++i) { mrow[i] = -3.0e38f; lrow[i] = 0.f; }
#pragma unroll
  for (int t = 0; t < 4; ++t)
#pragma unroll
    for (int i = 0; i < 8; ++i) o[t][i] = 0.f;

  for (int kblk = 0; kblk < 64; ++kblk) {
    const bf16_t* kbase = kb + (bh * 2048 + kblk * 32) * 64;
    __builtin_prefetch(kbase + 32 * 64, 0, 1);

    // S = Q @ K^T (two 16x16 tiles covering 32 keys)
    v8f s0, s1;
#pragma unroll
    for (int i = 0; i < 8; ++i) { s0[i] = 0.f; s1[i] = 0.f; }
    {
      AFrag b00 = load_b32x16(kbase, 64, lane);            // keys 0..15, d 0..31
      AFrag b01 = load_b32x16(kbase + 32, 64, lane);       // keys 0..15, d 32..63
      AFrag b10 = load_b32x16(kbase + 16 * 64, 64, lane);  // keys 16..31
      AFrag b11 = load_b32x16(kbase + 16 * 64 + 32, 64, lane);
      s0 = wmma_bf16(qa0, b00, s0); s0 = wmma_bf16(qa1, b01, s0);
      s1 = wmma_bf16(qa0, b10, s1); s1 = wmma_bf16(qa1, b11, s1);
    }

    // Online softmax; row m = i + 8*half lives entirely in this 16-lane half.
    float alpha[8];
#pragma unroll
    for (int i = 0; i < 8; ++i) {
      const float a = s0[i] * scale;
      const float b = s1[i] * scale;
      const float mx   = hmax16(fmaxf(a, b));
      const float mnew = fmaxf(mrow[i], mx);
      alpha[i] = __expf(mrow[i] - mnew);
      const float p0 = __expf(a - mnew);
      const float p1 = __expf(b - mnew);
      lrow[i] = lrow[i] * alpha[i] + hsum16(p0 + p1);
      mrow[i] = mnew;
      psh[w][i + 8 * half][n]      = (bf16_t)p0;
      psh[w][i + 8 * half][16 + n] = (bf16_t)p1;
    }
#pragma unroll
    for (int t = 0; t < 4; ++t)
#pragma unroll
      for (int i = 0; i < 8; ++i) o[t][i] *= alpha[i];

    asm volatile("s_wait_dscnt 0" ::: "memory");   // C-layout -> A-layout turnaround

    const AFrag pa = load_a16x32(&psh[w][0][0], 40, lane);  // P 16x32
    const bf16_t* vbase = vtb + (bh * 64) * 2048 + kblk * 32;
    __builtin_prefetch(vbase + 32, 0, 1);
#pragma unroll
    for (int t = 0; t < 4; ++t) {
      AFrag vb = load_b32x16(vbase + (16 * t) * 2048, 2048, lane);  // 32 keys x 16 dcols
      o[t] = wmma_bf16(pa, vb, o[t]);
    }
  }

  // Normalize and emit [B, N, dim] bf16 for the output projection.
  const int b = bh >> 4, h = bh & 15;
#pragma unroll
  for (int t = 0; t < 4; ++t)
#pragma unroll
    for (int i = 0; i < 8; ++i) {
      const int ns = r0 + i + 8 * half;
      const float val = o[t][i] / lrow[i];
      ab[(b * 2048 + ns) * 1024 + h * 64 + 16 * t + n] = (bf16_t)val;
    }
}

// ---------------------------------------------------------------------------
// GEMM 2: out = attn @ Wout + bout  (M=4096, K=1024, N=1024), f32 output.
// grid (128, 4), block 128
// ---------------------------------------------------------------------------
__global__ void __launch_bounds__(128) gemm_out(const bf16_t* __restrict__ ab,
                                                const bf16_t* __restrict__ wT,
                                                const float* __restrict__ bias,
                                                float* __restrict__ out) {
  const int lane = threadIdx.x & 31;
  const int w    = threadIdx.x >> 5;
  const int r0   = blockIdx.x * 32;
  const int c0   = blockIdx.y * 256 + w * 64;

  v8f acc[2][4];
#pragma unroll
  for (int mt = 0; mt < 2; ++mt)
#pragma unroll
    for (int nt = 0; nt < 4; ++nt)
#pragma unroll
      for (int i = 0; i < 8; ++i) acc[mt][nt][i] = 0.f;

  for (int k0 = 0; k0 < 1024; k0 += 32) {
    AFrag a0 = load_a16x32(ab + r0 * 1024 + k0, 1024, lane);
    AFrag a1 = load_a16x32(ab + (r0 + 16) * 1024 + k0, 1024, lane);
#pragma unroll
    for (int nt = 0; nt < 4; ++nt) {
      AFrag b = load_b32x16(wT + (c0 + 16 * nt) * 1024 + k0, 1024, lane);
      acc[0][nt] = wmma_bf16(a0, b, acc[0][nt]);
      acc[1][nt] = wmma_bf16(a1, b, acc[1][nt]);
    }
  }

  const int half = (lane >> 4) & 1, n = lane & 15;
#pragma unroll
  for (int mt = 0; mt < 2; ++mt)
#pragma unroll
    for (int nt = 0; nt < 4; ++nt) {
      const int col = c0 + 16 * nt + n;
      const float bv = bias[col];
#pragma unroll
      for (int i = 0; i < 8; ++i) {
        const int row = r0 + 16 * mt + i + 8 * half;
        out[row * 1024 + col] = acc[mt][nt][i] + bv;
      }
    }
}

// ---------------------------------------------------------------------------
// Launch: convert -> QKV GEMM -> flash attention -> output GEMM
// Workspace layout (bytes):
//   [0,8M)   xb   : x bf16 [4096,1024]   (reused as attention output `ab`)
//   [8M,14M) WqkvT: bf16 [3072,1024]
//   [14M,16M)WoutT: bf16 [1024,1024]
//   [16M,24M)qb   : bf16 [32,2048,64]
//   [24M,32M)kb   : bf16 [32,2048,64]
//   [32M,40M)vtb  : bf16 [32,64,2048]   (V transposed)
// ---------------------------------------------------------------------------
extern "C" void kernel_launch(void* const* d_in, const int* in_sizes, int n_in,
                              void* d_out, int out_size, void* d_ws, size_t ws_size,
                              hipStream_t stream) {
  const float* x    = (const float*)d_in[0];
  const float* Wqkv = (const float*)d_in[1];
  const float* bqkv = (const float*)d_in[2];
  const float* Wout = (const float*)d_in[3];
  const float* bout = (const float*)d_in[4];
  float* out = (float*)d_out;

  char* ws = (char*)d_ws;
  const size_t MB = 1u << 20;
  bf16_t* xb    = (bf16_t*)(ws);
  bf16_t* wqkvT = (bf16_t*)(ws + 8  * MB);
  bf16_t* woutT = (bf16_t*)(ws + 14 * MB);
  bf16_t* qb    = (bf16_t*)(ws + 16 * MB);
  bf16_t* kb    = (bf16_t*)(ws + 24 * MB);
  bf16_t* vtb   = (bf16_t*)(ws + 32 * MB);
  bf16_t* ab    = xb;  // x bf16 is dead after gemm_qkv; reuse for attention out

  // 1) data prep: f32 -> bf16 (+ weight transpose for contiguous B fragments)
  cvt_bf16  <<<dim3(16384), 256, 0, stream>>>(x, xb, 4096 * 1024);
  cvt_bf16_T<<<dim3(12288), 256, 0, stream>>>(Wqkv, wqkvT, 1024, 3072);
  cvt_bf16_T<<<dim3(4096),  256, 0, stream>>>(Wout, woutT, 1024, 1024);

  // 2) QKV projection -> q / k / v^T head-major buffers
  gemm_qkv<<<dim3(128, 12), 128, 0, stream>>>(xb, wqkvT, bqkv, qb, kb, vtb);

  // 3) flash attention (scale = sqrt(d_head) = 8, matching the reference)
  attn_fa<<<dim3(1024), 128, 0, stream>>>(qb, kb, vtb, ab, 8.0f);

  // 4) output projection (f32 out)
  gemm_out<<<dim3(128, 4), 128, 0, stream>>>(ab, woutT, bout, out);

  (void)in_sizes; (void)n_in; (void)out_size; (void)ws_size;
}
